// GAT_32263794328324
// MI455X (gfx1250) — matrix operand order repaired and checked by reading, not compiled, not run on hardware
//
#include <hip/hip_runtime.h>
#include <hip/hip_bf16.h>

// ---------------- problem constants (match reference) ----------------
#define NN      50000          // N_NODES
#define NE      800000         // N_EDGES
#define INF_    128            // IN_FEAT
#define HIDC    64             // HID (channels per head)
#define HEADS   4
#define HW      256            // HEADS*HID
#define OUTF    40
#define ETOT    (NE + NN)      // edges + self loops

typedef __attribute__((ext_vector_type(16))) __bf16 v16bf;
typedef __attribute__((ext_vector_type(8)))  float  v8f;

struct alignas(16) U4 { unsigned int a, b, c, d; };
union  Frag { U4 u[2]; v16bf v; };
struct alignas(16) F4 { float x, y, z, w; };

// float -> bf16 round-to-nearest-even (bit manipulation; no __bf16 scalar math)
__device__ __forceinline__ unsigned short f2bf(float f) {
    unsigned int u = __float_as_uint(f);
    u += 0x7FFFu + ((u >> 16) & 1u);
    return (unsigned short)(u >> 16);
}

__device__ __forceinline__ void atomicMaxF(float* addr, float val) {
    // standard monotone int/uint encoding trick; works with -inf init
    if (val >= 0.0f) atomicMax((int*)addr, __float_as_int(val));
    else             atomicMin((unsigned int*)addr, __float_as_uint(val));
}

// ---------------- elementwise / prep kernels ----------------
__global__ void k_cast_bf16(const float* __restrict__ in,
                            unsigned short* __restrict__ out, int n) {
    int t = blockIdx.x * blockDim.x + threadIdx.x;
    if (t < n) out[t] = f2bf(in[t]);
}

// W [K, Nout] f32 row-major  ->  Wt [Npad, K] bf16 (zero-padded columns)
__global__ void k_transpose_w(const float* __restrict__ W,
                              unsigned short* __restrict__ Wt,
                              int K, int Nout, int Npad) {
    int t = blockIdx.x * blockDim.x + threadIdx.x;
    if (t >= Npad * K) return;
    int n = t / K, k = t - n * K;
    float v = (n < Nout) ? W[(size_t)k * Nout + n] : 0.0f;
    Wt[(size_t)n * K + k] = f2bf(v);
}

__global__ void k_zero4(float* __restrict__ p, int n4) {
    int t = blockIdx.x * blockDim.x + threadIdx.x;
    if (t < n4) ((F4*)p)[t] = F4{0.f, 0.f, 0.f, 0.f};
}

// ---------------- bf16 WMMA GEMM: out = A[M,K] @ Bt[Npad,K]^T ----------------
// 8 waves/block as 2(M) x 4(N); each wave computes a 32x64 tile.
// K-loop software-pipelined depth 2 so loads overlap WMMAs.
__global__ __launch_bounds__(256) void k_gemm_bf16(
        const unsigned short* __restrict__ A,    // [M,K] bf16
        const unsigned short* __restrict__ Bt,   // [Npad,K] bf16 (transposed W)
        const float* __restrict__ bias,          // [Nreal] or null
        float* __restrict__ outF,                // [M,Nreal] f32 or null
        unsigned short* __restrict__ outB,       // [M,Nreal] bf16 or null
        int M, int K, int Npad, int Nreal, int relu)
{
    const int lane = threadIdx.x & 31;
    const int wave = threadIdx.x >> 5;
    const int mw = wave >> 2, nw = wave & 3;
    const int nstrips = Npad >> 6;
    const int strip = (nw < nstrips) ? nw : 0;
    const int rowBase = blockIdx.x * 64 + mw * 32;
    const int half = lane >> 4, l16 = lane & 15;
    const int colBase = strip << 6;

    // per-lane A row pointers (rows clamped; store guarded later)
    const unsigned short* arow[2];
#pragma unroll
    for (int rt = 0; rt < 2; ++rt) {
        int r = rowBase + rt * 16 + l16;
        if (r >= M) r = M - 1;
        arow[rt] = A + (size_t)r * K;
    }
    // per-lane B column pointers (cols always < Npad)
    const unsigned short* bcol[4];
#pragma unroll
    for (int t = 0; t < 4; ++t)
        bcol[t] = Bt + (size_t)(colBase + t * 16 + l16) * K;

    auto load_a = [&](Frag* fa, int k0) {
#pragma unroll
        for (int rt = 0; rt < 2; ++rt) {
            // ISA 16-bit A 16x32 layout: two contiguous 8-elem chunks per lane
            fa[rt].u[0] = *(const U4*)(arow[rt] + k0 + 8 * half);
            fa[rt].u[1] = *(const U4*)(arow[rt] + k0 + 16 + 8 * half);
        }
    };
    auto load_b = [&](Frag* fb, int k0) {
#pragma unroll
        for (int t = 0; t < 4; ++t) {
            // B 32x16 layout: lane = column, two 8-elem K chunks per lane (same as A)
            const unsigned short* bp = bcol[t] + k0 + 8 * half;
            fb[t].u[0] = *(const U4*)(bp);
            fb[t].u[1] = *(const U4*)(bp + 16);
        }
    };

    v8f acc[2][4] = {};
    Frag a0[2], b0[4], a1[2], b1[4];
    load_a(a0, 0); load_b(b0, 0);                    // prologue: stage 0

    for (int k0 = 0; k0 < K; k0 += 64) {             // K is a multiple of 64
        load_a(a1, k0 + 32); load_b(b1, k0 + 32);    // issue stage-1 loads
#pragma unroll
        for (int rt = 0; rt < 2; ++rt)
#pragma unroll
            for (int t = 0; t < 4; ++t)
                acc[rt][t] = __builtin_amdgcn_wmma_f32_16x16x32_bf16(
                    false, a0[rt].v, false, b0[t].v, (short)0,
                    acc[rt][t], false, false);
        if (k0 + 64 < K) { load_a(a0, k0 + 64); load_b(b0, k0 + 64); }
#pragma unroll
        for (int rt = 0; rt < 2; ++rt)
#pragma unroll
            for (int t = 0; t < 4; ++t)
                acc[rt][t] = __builtin_amdgcn_wmma_f32_16x16x32_bf16(
                    false, a1[rt].v, false, b1[t].v, (short)0,
                    acc[rt][t], false, false);
    }

    if (nw >= nstrips) return;
#pragma unroll
    for (int rt = 0; rt < 2; ++rt) {
#pragma unroll
        for (int t = 0; t < 4; ++t) {
#pragma unroll
            for (int r = 0; r < 8; ++r) {
                int row = rowBase + rt * 16 + half * 8 + r;  // C: VGPR r -> M=r(+8)
                int col = colBase + t * 16 + l16;
                if (row < M && col < Nreal) {
                    float v = acc[rt][t][r];
                    if (bias) v += bias[col];
                    if (relu) v = v > 0.f ? v : 0.f;
                    if (outF) outF[(size_t)row * Nreal + col] = v;
                    if (outB) outB[(size_t)row * Nreal + col] = f2bf(v);
                }
            }
        }
    }
}

// ---------------- GAT attention kernels ----------------
// per-(node,head): attention logits + init amax/-inf, den/0
__global__ __launch_bounds__(256) void k_logits(
        const float* __restrict__ hf, const float* __restrict__ as,
        const float* __restrict__ ad, float* __restrict__ sl,
        float* __restrict__ dl, float* __restrict__ amax,
        float* __restrict__ den)
{
    int t = blockIdx.x * blockDim.x + threadIdx.x;
    if (t >= NN * HEADS) return;
    int n = t >> 2, hh = t & 3;
    const float* hp = hf + (size_t)n * HW + hh * HIDC;
    const float* ap = as + hh * HIDC;
    const float* bp = ad + hh * HIDC;
    float sa = 0.f, da = 0.f;
#pragma unroll 4
    for (int i = 0; i < HIDC; ++i) { float hv = hp[i]; sa += hv * ap[i]; da += hv * bp[i]; }
    sl[t] = sa; dl[t] = da;
    amax[t] = -__builtin_inff();
    den[t] = 0.f;
}

__device__ __forceinline__ void edge_sd(int e, const long long* src,
                                        const long long* dst, int& s, int& d) {
    if (e < NE) { s = (int)src[e]; d = (int)dst[e]; }
    else        { s = d = e - NE; }                       // self loop
}

__global__ __launch_bounds__(256) void k_alpha_max(
        const long long* __restrict__ src, const long long* __restrict__ dst,
        const float* __restrict__ sl, const float* __restrict__ dl,
        float* __restrict__ amax)
{
    int t = blockIdx.x * blockDim.x + threadIdx.x;
    if (t >= ETOT * HEADS) return;
    int e = t >> 2, hh = t & 3, s, d;
    edge_sd(e, src, dst, s, d);
    float a = sl[s * 4 + hh] + dl[d * 4 + hh];
    a = a > 0.f ? a : 0.2f * a;                           // leaky_relu(0.2)
    atomicMaxF(&amax[d * 4 + hh], a);
}

__global__ __launch_bounds__(256) void k_expsum(
        const long long* __restrict__ src, const long long* __restrict__ dst,
        const float* __restrict__ sl, const float* __restrict__ dl,
        const float* __restrict__ amax, float* __restrict__ den)
{
    int t = blockIdx.x * blockDim.x + threadIdx.x;
    if (t >= ETOT * HEADS) return;
    int e = t >> 2, hh = t & 3, s, d;
    edge_sd(e, src, dst, s, d);
    float a = sl[s * 4 + hh] + dl[d * 4 + hh];
    a = a > 0.f ? a : 0.2f * a;
    float ex = __expf(a - amax[d * 4 + hh]);
    __hip_atomic_fetch_add(&den[d * 4 + hh], ex,
                           __ATOMIC_RELAXED, __HIP_MEMORY_SCOPE_AGENT);
}

// one wave per edge; lane handles 8 consecutive channels (single head per lane)
__global__ __launch_bounds__(256) void k_message(
        const long long* __restrict__ src, const long long* __restrict__ dst,
        const float* __restrict__ sl, const float* __restrict__ dl,
        const float* __restrict__ amax, const float* __restrict__ den,
        const float* __restrict__ hf, float* __restrict__ agg)
{
    int gid = blockIdx.x * blockDim.x + threadIdx.x;
    int e = gid >> 5;
    if (e >= ETOT) return;
    int lane = gid & 31, s, d;
    edge_sd(e, src, dst, s, d);
    int c0 = lane * 8;
    int hh = c0 >> 6;
    float a = sl[s * 4 + hh] + dl[d * 4 + hh];
    a = a > 0.f ? a : 0.2f * a;
    float coef = __expf(a - amax[d * 4 + hh]) / den[d * 4 + hh];

    const float* hp = hf + (size_t)s * HW + c0;           // L2-resident gather
    float* op = agg + (size_t)d * HW + c0;
    F4 v0 = *(const F4*)(hp);
    F4 v1 = *(const F4*)(hp + 4);
    __hip_atomic_fetch_add(op + 0, v0.x * coef, __ATOMIC_RELAXED, __HIP_MEMORY_SCOPE_AGENT);
    __hip_atomic_fetch_add(op + 1, v0.y * coef, __ATOMIC_RELAXED, __HIP_MEMORY_SCOPE_AGENT);
    __hip_atomic_fetch_add(op + 2, v0.z * coef, __ATOMIC_RELAXED, __HIP_MEMORY_SCOPE_AGENT);
    __hip_atomic_fetch_add(op + 3, v0.w * coef, __ATOMIC_RELAXED, __HIP_MEMORY_SCOPE_AGENT);
    __hip_atomic_fetch_add(op + 4, v1.x * coef, __ATOMIC_RELAXED, __HIP_MEMORY_SCOPE_AGENT);
    __hip_atomic_fetch_add(op + 5, v1.y * coef, __ATOMIC_RELAXED, __HIP_MEMORY_SCOPE_AGENT);
    __hip_atomic_fetch_add(op + 6, v1.z * coef, __ATOMIC_RELAXED, __HIP_MEMORY_SCOPE_AGENT);
    __hip_atomic_fetch_add(op + 7, v1.w * coef, __ATOMIC_RELAXED, __HIP_MEMORY_SCOPE_AGENT);
}

// out-of-conv bias + relu + cast to bf16 (next layer's GEMM input)
__global__ __launch_bounds__(256) void k_finalize(
        const float* __restrict__ agg, const float* __restrict__ bias,
        unsigned short* __restrict__ hb)
{
    int t = blockIdx.x * blockDim.x + threadIdx.x;
    if (t >= NN * HW) return;
    int c = t & (HW - 1);
    float v = agg[t] + bias[c];
    v = v > 0.f ? v : 0.f;
    hb[t] = f2bf(v);
}

// ---------------- host launcher ----------------
static inline int ce(long long a, long long b) { return (int)((a + b - 1) / b); }

extern "C" void kernel_launch(void* const* d_in, const int* in_sizes, int n_in,
                              void* d_out, int out_size, void* d_ws, size_t ws_size,
                              hipStream_t stream) {
    (void)in_sizes; (void)n_in; (void)out_size; (void)ws_size;
    const float*      x   = (const float*)d_in[0];
    const long long*  ei  = (const long long*)d_in[1];    // int64 [2, NE]
    const long long*  src = ei;
    const long long*  dst = ei + NE;
    const float *W1 = (const float*)d_in[2],  *as1 = (const float*)d_in[3],
                *ad1 = (const float*)d_in[4], *b1  = (const float*)d_in[5];
    const float *W2 = (const float*)d_in[6],  *as2 = (const float*)d_in[7],
                *ad2 = (const float*)d_in[8], *b2  = (const float*)d_in[9];
    const float *W3 = (const float*)d_in[10], *as3 = (const float*)d_in[11],
                *ad3 = (const float*)d_in[12],*b3  = (const float*)d_in[13];
    const float *Wm1 = (const float*)d_in[14], *bm1 = (const float*)d_in[15];
    const float *Wm2 = (const float*)d_in[16], *bm2 = (const float*)d_in[17];

    // bump allocator over workspace
    char* p = (char*)d_ws;
    auto alloc = [&](size_t bytes) -> void* {
        void* r = (void*)p; p += (bytes + 255) & ~(size_t)255; return r;
    };
    unsigned short* xb   = (unsigned short*)alloc((size_t)NN * INF_ * 2);
    unsigned short* hb   = (unsigned short*)alloc((size_t)NN * HW * 2);
    float*          hf   = (float*)alloc((size_t)NN * HW * 4);
    float*          agg  = (float*)alloc((size_t)NN * HW * 4);
    unsigned short* hb2  = (unsigned short*)hf;           // alias: hf dead before MLP1
    unsigned short* W1t  = (unsigned short*)alloc((size_t)HW * INF_ * 2);
    unsigned short* W2t  = (unsigned short*)alloc((size_t)HW * HW * 2);
    unsigned short* W3t  = (unsigned short*)alloc((size_t)HW * HW * 2);
    unsigned short* Wm1t = (unsigned short*)alloc((size_t)HW * HW * 2);
    unsigned short* Wm2t = (unsigned short*)alloc((size_t)64 * HW * 2);
    float* sl   = (float*)alloc((size_t)NN * HEADS * 4);
    float* dl   = (float*)alloc((size_t)NN * HEADS * 4);
    float* amax = (float*)alloc((size_t)NN * HEADS * 4);
    float* den  = (float*)alloc((size_t)NN * HEADS * 4);

    const int gM = ce(NN, 64);                 // 782 blocks (64 rows each)

    // prep: cast inputs / transpose weights to bf16
    k_cast_bf16<<<ce((long long)NN * INF_, 256), 256, 0, stream>>>(x, xb, NN * INF_);
    k_transpose_w<<<ce(HW * INF_, 256), 256, 0, stream>>>(W1,  W1t,  INF_, HW,  HW);
    k_transpose_w<<<ce(HW * HW,  256), 256, 0, stream>>>(W2,  W2t,  HW,   HW,  HW);
    k_transpose_w<<<ce(HW * HW,  256), 256, 0, stream>>>(W3,  W3t,  HW,   HW,  HW);
    k_transpose_w<<<ce(HW * HW,  256), 256, 0, stream>>>(Wm1, Wm1t, HW,   HW,  HW);
    k_transpose_w<<<ce(64 * HW,  256), 256, 0, stream>>>(Wm2, Wm2t, HW, OUTF,  64);

    auto gat_layer = [&](const unsigned short* in_bf, int K,
                         const unsigned short* Wt, const float* as,
                         const float* ad, const float* bias) {
        k_gemm_bf16<<<gM, 256, 0, stream>>>(in_bf, Wt, nullptr, hf, nullptr,
                                            NN, K, HW, HW, 0);
        k_logits<<<ce((long long)NN * HEADS, 256), 256, 0, stream>>>(
            hf, as, ad, sl, dl, amax, den);
        k_zero4<<<ce((long long)NN * HW / 4, 256), 256, 0, stream>>>(agg, NN * HW / 4);
        k_alpha_max<<<ce((long long)ETOT * HEADS, 256), 256, 0, stream>>>(
            src, dst, sl, dl, amax);
        k_expsum<<<ce((long long)ETOT * HEADS, 256), 256, 0, stream>>>(
            src, dst, sl, dl, amax, den);
        k_message<<<ce((long long)ETOT * 32, 256), 256, 0, stream>>>(
            src, dst, sl, dl, amax, den, hf, agg);
        k_finalize<<<ce((long long)NN * HW, 256), 256, 0, stream>>>(agg, bias, hb);
    };

    gat_layer(xb, INF_, W1t, as1, ad1, b1);
    gat_layer(hb, HW,   W2t, as2, ad2, b2);
    gat_layer(hb, HW,   W3t, as3, ad3, b3);

    // MLP head: relu(h @ Wm1 + bm1) @ Wm2 + bm2
    k_gemm_bf16<<<gM, 256, 0, stream>>>(hb,  Wm1t, bm1, nullptr, hb2,
                                        NN, HW, HW, HW, 1);
    k_gemm_bf16<<<gM, 256, 0, stream>>>(hb2, Wm2t, bm2, (float*)d_out, nullptr,
                                        NN, HW, 64, OUTF, 0);
}
